// GAT_29326036697830
// MI455X (gfx1250) — matrix operand order repaired and checked by reading, not compiled, not run on hardware
//
#include <hip/hip_runtime.h>
#include <math.h>

// ---------------------------------------------------------------------------
// GAT pipeline for MI455X (gfx1250, wave32).
// GEMMs: v_wmma_f32_16x16x32_bf16; bf16 A (activations) and bf16 B^T (weights)
// are pre-converted once, then staged into LDS with b128 loads/stores.
// Edge softmax/aggregation: ordered-int atomicMax + float atomicAdd; the
// 102 MB feature matrix stays L2-resident (192 MB L2) for the random gathers.
// ---------------------------------------------------------------------------

typedef __attribute__((ext_vector_type(16))) __bf16 v16bf;
typedef __attribute__((ext_vector_type(16))) unsigned short v16u;
typedef __attribute__((ext_vector_type(8)))  float v8f;

#define N_NODES   100000
#define N_EDGES   400000
#define N_GRAPHS  4096
#define NEG_SLOPE 0.2f

__device__ __forceinline__ unsigned short f32_to_bf16(float f) {
  unsigned int u = __float_as_uint(f);
  u += 0x7fffu + ((u >> 16) & 1u);          // round-to-nearest-even
  return (unsigned short)(u >> 16);
}
__device__ __forceinline__ int ordered_of_float(float f) {
  int i = __float_as_int(f);
  return (i >= 0) ? i : (i ^ 0x7fffffff);   // monotone float->int map
}
__device__ __forceinline__ float float_of_ordered(int o) {
  return __int_as_float((o >= 0) ? o : (o ^ 0x7fffffff));
}

// ---------------------------------------------------------------------------
// WMMA GEMM: C[M,N] = A[M,K] @ B[K,N] with A bf16 row-major [M,K] and
// B supplied pre-transposed bf16 [N,K]. f32 accumulate, f32 output.
// Block = 128 threads (4 waves). Block tile 64x64, wave tile 16x64,
// K-chunks of 32. LDS rows padded to 40 ushorts (80 B) so every fragment
// read and staging store is a 16-byte-aligned b128 access.
// ---------------------------------------------------------------------------
__global__ __launch_bounds__(128)
void wmma_gemm_bf16(const unsigned short* __restrict__ A,
                    const unsigned short* __restrict__ BT,
                    float* __restrict__ C, int M, int K, int N) {
  __shared__ unsigned short As[64 * 40];   // [row][k]
  __shared__ unsigned short Bs[64 * 40];   // [col][k]

  const int tid  = threadIdx.x;
  const int wave = tid >> 5;
  const int lane = tid & 31;
  const int half = lane >> 4;
  const int l16  = lane & 15;
  const int m0   = blockIdx.x * 64;
  const int n0   = blockIdx.y * 64;

  v8f acc[4];
  #pragma unroll
  for (int i = 0; i < 4; ++i) acc[i] = v8f{};

  // Per-thread staging coordinates: tile = 64 rows x 4 quads (uint4 = 8 bf16).
  const int sr0 = tid >> 2;          // rows 0..31 (pass 0) / 32..63 (pass 1)
  const int sq  = tid & 3;           // quad within row

  for (int k0 = 0; k0 < K; k0 += 32) {
    // Stage A tile (64x32 bf16): two b128 loads/stores per thread.
    #pragma unroll
    for (int i = 0; i < 2; ++i) {
      int r  = sr0 + i * 32;
      int gr = m0 + r;
      uint4 v = {0u, 0u, 0u, 0u};
      if (gr < M) v = *(const uint4*)(A + (size_t)gr * K + k0 + sq * 8);
      *(uint4*)(As + r * 40 + sq * 8) = v;
    }
    // Stage B^T tile (64 cols x 32 k): two b128 loads/stores per thread.
    #pragma unroll
    for (int i = 0; i < 2; ++i) {
      int c = sr0 + i * 32;
      uint4 v = *(const uint4*)(BT + (size_t)(n0 + c) * K + k0 + sq * 8);
      *(uint4*)(Bs + c * 40 + sq * 8) = v;
    }
    // Prefetch next K-chunk (global_prefetch_b8).
    if (k0 + 32 < K) {
      int gr = m0 + sr0;
      if (gr < M) __builtin_prefetch(A + (size_t)gr * K + k0 + 32, 0, 0);
      __builtin_prefetch(BT + (size_t)(n0 + sr0) * K + k0 + 32, 0, 0);
    }
    __syncthreads();

    // A fragment (ISA 7.12.2: lane=row, half 0 -> K{0..7,16..23}, half 1 -> +8)
    const uint4* pa = (const uint4*)(As + (wave * 16 + l16) * 40);
    union { uint4 q[2]; v16u v; } ua;
    ua.q[0] = pa[half];
    ua.q[1] = pa[2 + half];
    v16bf afrag = __builtin_bit_cast(v16bf, ua.v);

    #pragma unroll
    for (int nt = 0; nt < 4; ++nt) {
      // B fragment: lane=col, half 0 -> K{0..7,16..23}, half 1 -> +8
      const uint4* pb = (const uint4*)(Bs + (nt * 16 + l16) * 40);
      union { uint4 q[2]; v16u v; } ub;
      ub.q[0] = pb[half];
      ub.q[1] = pb[2 + half];
      v16bf bfrag = __builtin_bit_cast(v16bf, ub.v);
      acc[nt] = __builtin_amdgcn_wmma_f32_16x16x32_bf16(
          false, afrag, false, bfrag, (short)0, acc[nt], false, false);
    }
    __syncthreads();
  }

  // C/D layout: VGPR r -> M = r (lanes 0-15) / 8+r (lanes 16-31), N = lane&15.
  #pragma unroll
  for (int nt = 0; nt < 4; ++nt) {
    int col = n0 + nt * 16 + l16;
    #pragma unroll
    for (int r = 0; r < 8; ++r) {
      int row = m0 + wave * 16 + half * 8 + r;
      if (row < M) C[(size_t)row * N + col] = acc[nt][r];
    }
  }
}

// ---------------------------------------------------------------------------
// One-shot converters.
// ---------------------------------------------------------------------------
__global__ void cvt_f32_bf16_kernel(const float* __restrict__ in,
                                    unsigned short* __restrict__ out, int n) {
  int t = blockIdx.x * blockDim.x + threadIdx.x;
  if (t < n) out[t] = f32_to_bf16(in[t]);
}

// WT[n*K + k] = bf16(W[k*N + n])  (weight transpose + convert; tiny)
__global__ void transpose_cvt_kernel(const float* __restrict__ W,
                                     unsigned short* __restrict__ WT,
                                     int K, int N) {
  int t = blockIdx.x * blockDim.x + threadIdx.x;
  if (t >= K * N) return;
  int k = t / N, n = t % N;
  WT[(size_t)n * K + k] = f32_to_bf16(W[t]);
}

// ---------------------------------------------------------------------------
// Attention coefficients: e_src[n,h] = <h[n,h,:], a_s[h,:]>, same for e_dst.
// ---------------------------------------------------------------------------
__global__ void attn_kernel(const float* __restrict__ h,
                            const float* __restrict__ a_s,
                            const float* __restrict__ a_d,
                            float* __restrict__ esrc, float* __restrict__ edst,
                            int Nn, int H, int Cc) {
  int t = blockIdx.x * blockDim.x + threadIdx.x;
  if (t >= Nn * H) return;
  int n = t / H, hd = t % H;
  const float* hp  = h + (size_t)n * H * Cc + hd * Cc;
  const float* asp = a_s + hd * Cc;
  const float* adp = a_d + hd * Cc;
  float ss = 0.f, sd = 0.f;
  for (int c = 0; c < Cc; ++c) { float v = hp[c]; ss += v * asp[c]; sd += v * adp[c]; }
  esrc[t] = ss; edst[t] = sd;
}

__global__ void fill_f32(float* __restrict__ p, float v, int n) {
  int t = blockIdx.x * blockDim.x + threadIdx.x;
  if (t < n) p[t] = v;
}
__global__ void fill_i32(int* __restrict__ p, int v, int n) {
  int t = blockIdx.x * blockDim.x + threadIdx.x;
  if (t < n) p[t] = v;
}

__device__ __forceinline__ void edge_decode(const int* __restrict__ ei, int e,
                                            int& s, int& d) {
  if (e < N_EDGES) { s = ei[e]; d = ei[N_EDGES + e]; }
  else             { s = d = e - N_EDGES; }        // synthesized self-loop
}

// Pass 1: segment max of leaky_relu(e_src[s]+e_dst[d]) per dst, ordered-int.
__global__ void segmax_kernel(const int* __restrict__ ei,
                              const float* __restrict__ esrc,
                              const float* __restrict__ edst,
                              int* __restrict__ mbuf, int H) {
  int e = blockIdx.x * blockDim.x + threadIdx.x;
  const int Et = N_EDGES + N_NODES;
  if (e >= Et) return;
  int s, d; edge_decode(ei, e, s, d);
  for (int hd = 0; hd < H; ++hd) {
    float v = esrc[s * H + hd] + edst[d * H + hd];
    v = (v > 0.f) ? v : NEG_SLOPE * v;
    atomicMax(&mbuf[d * H + hd], ordered_of_float(v));
  }
}

// Pass 2: denom[d,h] += exp(e - m[d,h])
__global__ void segsum_kernel(const int* __restrict__ ei,
                              const float* __restrict__ esrc,
                              const float* __restrict__ edst,
                              const int* __restrict__ mbuf,
                              float* __restrict__ denom, int H) {
  int e = blockIdx.x * blockDim.x + threadIdx.x;
  const int Et = N_EDGES + N_NODES;
  if (e >= Et) return;
  int s, d; edge_decode(ei, e, s, d);
  for (int hd = 0; hd < H; ++hd) {
    float v = esrc[s * H + hd] + edst[d * H + hd];
    v = (v > 0.f) ? v : NEG_SLOPE * v;
    float m = float_of_ordered(mbuf[d * H + hd]);
    atomicAdd(&denom[d * H + hd], expf(v - m));
  }
}

// Pass 3: out[d,:] += alpha * h[s,:]   (one wave per edge, lane-strided feats)
__global__ void aggregate_kernel(const int* __restrict__ ei,
                                 const float* __restrict__ esrc,
                                 const float* __restrict__ edst,
                                 const int* __restrict__ mbuf,
                                 const float* __restrict__ denom,
                                 const float* __restrict__ h,
                                 float* __restrict__ out, int H, int Cc) {
  int gw = (int)((blockIdx.x * (size_t)blockDim.x + threadIdx.x) >> 5);
  int lane = threadIdx.x & 31;
  const int Et = N_EDGES + N_NODES;
  if (gw >= Et) return;
  int s, d; edge_decode(ei, gw, s, d);
  float alpha[4];
  for (int hd = 0; hd < H; ++hd) {
    float v = esrc[s * H + hd] + edst[d * H + hd];
    v = (v > 0.f) ? v : NEG_SLOPE * v;
    float m  = float_of_ordered(mbuf[d * H + hd]);
    float ex = expf(v - m);
    alpha[hd] = ex / fmaxf(denom[d * H + hd], 1e-16f);
  }
  int F = H * Cc;
  for (int f = lane; f < F; f += 32) {
    int hd = f / Cc;
    atomicAdd(&out[(size_t)d * F + f], alpha[hd] * h[(size_t)s * F + f]);
  }
}

// Finalize for layers 0-2: bias + ELU, emit bf16 activation for next GEMM.
__global__ void bias_elu_cvt_kernel(const float* __restrict__ in,
                                    const float* __restrict__ bias,
                                    unsigned short* __restrict__ outbf,
                                    int total, int F) {
  int t = blockIdx.x * blockDim.x + threadIdx.x;
  if (t >= total) return;
  float v = in[t] + bias[t % F];
  v = (v > 0.f) ? v : (expf(v) - 1.f);
  outbf[t] = f32_to_bf16(v);
}

// Finalize for layer 3: bias only, in-place f32 (consumed by pooling).
__global__ void bias_kernel(float* __restrict__ out,
                            const float* __restrict__ bias, int total, int F) {
  int t = blockIdx.x * blockDim.x + threadIdx.x;
  if (t >= total) return;
  out[t] = out[t] + bias[t % F];
}

__global__ void pool_kernel(const float* __restrict__ h,
                            const int* __restrict__ batch,
                            float* __restrict__ pooled, float* __restrict__ cnt,
                            int F) {
  int t = blockIdx.x * blockDim.x + threadIdx.x;
  if (t >= N_NODES * F) return;
  int n = t / F, f = t % F;
  int g = batch[n];
  atomicAdd(&pooled[(size_t)g * F + f], h[t]);
  if (f == 0) atomicAdd(&cnt[g], 1.0f);
}

// MLPReadout: 128 -> 64 -> 32 -> 4 with ReLU. One block per graph.
__global__ __launch_bounds__(128)
void mlp_kernel(const float* __restrict__ pooled, const float* __restrict__ cnt,
                const float* __restrict__ mW0, const float* __restrict__ mb0,
                const float* __restrict__ mW1, const float* __restrict__ mb1,
                const float* __restrict__ mW2, const float* __restrict__ mb2,
                float* __restrict__ y) {
  __shared__ float p[128];
  __shared__ float t0[64];
  __shared__ float t1[32];
  int g = blockIdx.x, tid = threadIdx.x;
  float c = fmaxf(cnt[g], 1.0f);
  p[tid] = pooled[(size_t)g * 128 + tid] / c;
  __syncthreads();
  if (tid < 64) {
    float s = mb0[tid];
    for (int k = 0; k < 128; ++k) s += p[k] * mW0[k * 64 + tid];
    t0[tid] = fmaxf(s, 0.f);
  }
  __syncthreads();
  if (tid < 32) {
    float s = mb1[tid];
    for (int k = 0; k < 64; ++k) s += t0[k] * mW1[k * 32 + tid];
    t1[tid] = fmaxf(s, 0.f);
  }
  __syncthreads();
  if (tid < 4) {
    float s = mb2[tid];
    for (int k = 0; k < 32; ++k) s += t1[k] * mW2[k * 4 + tid];
    y[(size_t)g * 4 + tid] = s;
  }
}

// ---------------------------------------------------------------------------
// Host-side orchestration (all launches on `stream`; graph-capture safe).
// ---------------------------------------------------------------------------
static void run_gat_layer(const unsigned short* actbf, int K,
                          const float* W, const float* a_s, const float* a_d,
                          const float* b, int H, int Cc, int out_bf16,
                          const int* ei, unsigned short* wT,
                          float* hbuf, float* outbuf, unsigned short* actbf_out,
                          float* esrc, float* edst, int* mbuf, float* denom,
                          hipStream_t stream) {
  const int F  = H * Cc;
  const int nh = N_NODES * H;
  const int nf = N_NODES * F;
  const int Et = N_EDGES + N_NODES;

  transpose_cvt_kernel<<<(K * F + 255) / 256, 256, 0, stream>>>(W, wT, K, F);

  dim3 gg((N_NODES + 63) / 64, F / 64);
  wmma_gemm_bf16<<<gg, 128, 0, stream>>>(actbf, wT, hbuf, N_NODES, K, F);

  attn_kernel<<<(nh + 255) / 256, 256, 0, stream>>>(hbuf, a_s, a_d, esrc, edst,
                                                    N_NODES, H, Cc);
  fill_i32<<<(nh + 255) / 256, 256, 0, stream>>>(mbuf, (int)0x80000000, nh);
  fill_f32<<<(nh + 255) / 256, 256, 0, stream>>>(denom, 0.f, nh);
  fill_f32<<<(nf + 255) / 256, 256, 0, stream>>>(outbuf, 0.f, nf);

  segmax_kernel<<<(Et + 255) / 256, 256, 0, stream>>>(ei, esrc, edst, mbuf, H);
  segsum_kernel<<<(Et + 255) / 256, 256, 0, stream>>>(ei, esrc, edst, mbuf, denom, H);
  aggregate_kernel<<<((size_t)Et * 32 + 255) / 256, 256, 0, stream>>>(
      ei, esrc, edst, mbuf, denom, hbuf, outbuf, H, Cc);

  if (out_bf16)
    bias_elu_cvt_kernel<<<(nf + 255) / 256, 256, 0, stream>>>(outbuf, b,
                                                              actbf_out, nf, F);
  else
    bias_kernel<<<(nf + 255) / 256, 256, 0, stream>>>(outbuf, b, nf, F);
}

extern "C" void kernel_launch(void* const* d_in, const int* in_sizes, int n_in,
                              void* d_out, int out_size, void* d_ws, size_t ws_size,
                              hipStream_t stream) {
  (void)in_sizes; (void)n_in; (void)out_size; (void)ws_size;

  const float* x     = (const float*)d_in[0];
  const int*   ei    = (const int*)  d_in[1];
  const int*   batch = (const int*)  d_in[2];
  // d_in[3] = num_graphs (compile-time constant N_GRAPHS)
  const float* W0  = (const float*)d_in[4];
  const float* as0 = (const float*)d_in[5];
  const float* ad0 = (const float*)d_in[6];
  const float* b0  = (const float*)d_in[7];
  const float* W1  = (const float*)d_in[8];
  const float* as1 = (const float*)d_in[9];
  const float* ad1 = (const float*)d_in[10];
  const float* b1  = (const float*)d_in[11];
  const float* W2  = (const float*)d_in[12];
  const float* as2 = (const float*)d_in[13];
  const float* ad2 = (const float*)d_in[14];
  const float* b2  = (const float*)d_in[15];
  const float* W3  = (const float*)d_in[16];
  const float* as3 = (const float*)d_in[17];
  const float* ad3 = (const float*)d_in[18];
  const float* b3  = (const float*)d_in[19];
  const float* mW0 = (const float*)d_in[20];
  const float* mb0 = (const float*)d_in[21];
  const float* mW1 = (const float*)d_in[22];
  const float* mb1 = (const float*)d_in[23];
  const float* mW2 = (const float*)d_in[24];
  const float* mb2 = (const float*)d_in[25];

  // Workspace carve-up (~263 MB, all chunks 16B-aligned).
  char* w = (char*)d_ws;
  float*          hbuf   = (float*)w;          w += (size_t)N_NODES * 256 * sizeof(float);
  float*          outb   = (float*)w;          w += (size_t)N_NODES * 256 * sizeof(float);
  unsigned short* actbf  = (unsigned short*)w; w += (size_t)N_NODES * 256 * sizeof(unsigned short);
  unsigned short* wT     = (unsigned short*)w; w += (size_t)256 * 256 * sizeof(unsigned short);
  float*          esrc   = (float*)w;          w += (size_t)N_NODES * 4 * sizeof(float);
  float*          edst   = (float*)w;          w += (size_t)N_NODES * 4 * sizeof(float);
  int*            mbuf   = (int*)w;            w += (size_t)N_NODES * 4 * sizeof(int);
  float*          denom  = (float*)w;          w += (size_t)N_NODES * 4 * sizeof(float);
  float*          pooled = (float*)w;          w += (size_t)N_GRAPHS * 128 * sizeof(float);
  float*          cnt    = (float*)w;          w += (size_t)N_GRAPHS * sizeof(float);

  // x -> bf16 (layer 0 GEMM input, [N,64])
  cvt_f32_bf16_kernel<<<(N_NODES * 64 + 255) / 256, 256, 0, stream>>>(
      x, actbf, N_NODES * 64);

  // Layer 0: actbf[N,64] -> actbf[N,256] (elu, bf16 out)
  run_gat_layer(actbf, 64,  W0, as0, ad0, b0, 4, 64, 1, ei, wT,
                hbuf, outb, actbf, esrc, edst, mbuf, denom, stream);
  // Layers 1,2: actbf[N,256] -> actbf[N,256]
  run_gat_layer(actbf, 256, W1, as1, ad1, b1, 4, 64, 1, ei, wT,
                hbuf, outb, actbf, esrc, edst, mbuf, denom, stream);
  run_gat_layer(actbf, 256, W2, as2, ad2, b2, 4, 64, 1, ei, wT,
                hbuf, outb, actbf, esrc, edst, mbuf, denom, stream);
  // Layer 3: H=1, C=128, concat=False (mean over 1 head == identity), f32 out
  run_gat_layer(actbf, 256, W3, as3, ad3, b3, 1, 128, 0, ei, wT,
                hbuf, outb, actbf, esrc, edst, mbuf, denom, stream);

  // Global mean pool per graph (reads layer-3 f32 output in outb)
  fill_f32<<<(N_GRAPHS * 128 + 255) / 256, 256, 0, stream>>>(pooled, 0.f, N_GRAPHS * 128);
  fill_f32<<<(N_GRAPHS + 255) / 256, 256, 0, stream>>>(cnt, 0.f, N_GRAPHS);
  pool_kernel<<<((size_t)N_NODES * 128 + 255) / 256, 256, 0, stream>>>(
      outb, batch, pooled, cnt, 128);

  // MLP readout 128 -> 64 -> 32 -> 4
  mlp_kernel<<<N_GRAPHS, 128, 0, stream>>>(pooled, cnt, mW0, mb0, mW1, mb1,
                                           mW2, mb2, (float*)d_out);
}
